// Head_4320737099973
// MI455X (gfx1250) — matrix operand order repaired and checked by reading, not compiled, not run on hardware
//
#include <hip/hip_runtime.h>
#include <hip/hip_bf16.h>

typedef __attribute__((ext_vector_type(16))) __bf16 v16bf;
typedef __attribute__((ext_vector_type(8)))  __bf16 v8bf;
typedef __attribute__((ext_vector_type(8)))  float  v8f;
typedef __attribute__((ext_vector_type(4)))  float  f4;

#define TSEQ 2048
#define CDIM 1024
#define HDIM 64
#define NB   8

static __device__ __forceinline__ v8f wmma_bf16(v16bf a, v16bf b, v8f c) {
  // D = A(16x32 bf16) x B(32x16 bf16) + C(16x16 f32)
  return __builtin_amdgcn_wmma_f32_16x16x32_bf16(
      /*neg_a=*/false, a, /*neg_b=*/false, b,
      /*c_mod=*/(short)0, c, /*reuse_a=*/false, /*reuse_b=*/false);
}

// ---------------------------------------------------------------------------
// Kernel 1: convert + transpose weights: Wt[p][h][c] (bf16) = W_p[c][h] (f32)
// ---------------------------------------------------------------------------
__global__ __launch_bounds__(256)
void wconvert(const float* __restrict__ Wq, const float* __restrict__ Wk,
              const float* __restrict__ Wv, __bf16* __restrict__ Wt) {
  int idx = blockIdx.x * 256 + threadIdx.x;   // 3*64*1024 total
  int p   = idx >> 16;
  int rem = idx & 65535;
  int h   = rem >> 10;
  int c   = rem & 1023;
  const float* W = (p == 0) ? Wq : ((p == 1) ? Wk : Wv);
  Wt[(size_t)p * (HDIM * CDIM) + (size_t)h * CDIM + c] = (__bf16)W[c * HDIM + h];
}

// ---------------------------------------------------------------------------
// Kernel 2: QKV projection. Block = 16 rows of x, 4 waves (one 16-col tile
// each, shared across Q/K/V). Writes Qb,Kb as bf16 [B*T][64], V transposed
// as Vt bf16 [B][64][T].
// ---------------------------------------------------------------------------
__global__ __launch_bounds__(128)
void qkv_proj(const float* __restrict__ x, const __bf16* __restrict__ Wt,
              __bf16* __restrict__ Qb, __bf16* __restrict__ Kb,
              __bf16* __restrict__ Vt) {
  const int rbase = blockIdx.x * 16;          // global row (in B*T)
  const int wave  = threadIdx.x >> 5;         // 0..3 => N tile (h cols)
  const int lane  = threadIdx.x & 31;
  const int l16   = lane & 15;
  const int half  = lane >> 4;

  v8f accQ = {}, accK = {}, accV = {};
  const float* xrow = x + (size_t)(rbase + l16) * CDIM;
  const int hcol = wave * 16 + l16;

  for (int kc = 0; kc < CDIM; kc += 32) {
    // A operand: 16x32 f32 tile of x, converted to bf16, contiguous K layout
    v16bf a;
    {
      const float* pa = xrow + kc + half * 16;
      f4 x0 = *(const f4*)(pa);
      f4 x1 = *(const f4*)(pa + 4);
      f4 x2 = *(const f4*)(pa + 8);
      f4 x3 = *(const f4*)(pa + 12);
      #pragma unroll
      for (int e = 0; e < 4; ++e) {
        a[e]      = (__bf16)x0[e];
        a[4 + e]  = (__bf16)x1[e];
        a[8 + e]  = (__bf16)x2[e];
        a[12 + e] = (__bf16)x3[e];
      }
    }
    // B operands: contiguous 16 halves of Wt row (lane = N, K = half*16+e)
    const size_t wof = (size_t)hcol * CDIM + kc + half * 16;
    v16bf bq = *(const v16bf*)(Wt + 0 * (HDIM * CDIM) + wof);
    v16bf bk = *(const v16bf*)(Wt + 1 * (HDIM * CDIM) + wof);
    v16bf bv = *(const v16bf*)(Wt + 2 * (HDIM * CDIM) + wof);
    accQ = wmma_bf16(a, bq, accQ);
    accK = wmma_bf16(a, bk, accK);
    accV = wmma_bf16(a, bv, accV);
  }

  const int b  = rbase >> 11;           // / TSEQ
  const int t0 = rbase & (TSEQ - 1);
  #pragma unroll
  for (int r = 0; r < 8; ++r) {
    const int row = rbase + r + 8 * half;           // D layout: M = r + 8*half
    Qb[(size_t)row * HDIM + hcol] = (__bf16)accQ[r];
    Kb[(size_t)row * HDIM + hcol] = (__bf16)accK[r];
    Vt[(size_t)b * HDIM * TSEQ + (size_t)hcol * TSEQ + (t0 + r + 8 * half)] =
        (__bf16)accV[r];
  }
}

// ---------------------------------------------------------------------------
// Kernel 3: causal flash attention. 256 threads = 8 waves; each wave owns 16
// query rows, streams 32-key tiles up to the diagonal with online softmax.
// ---------------------------------------------------------------------------
__global__ __launch_bounds__(256)
void attn(const __bf16* __restrict__ Qb, const __bf16* __restrict__ Kb,
          const __bf16* __restrict__ Vt, float* __restrict__ out) {
  __shared__ __bf16 pbuf[8][16 * 40];   // per-wave P staging, padded rows

  const int wave = threadIdx.x >> 5;
  const int lane = threadIdx.x & 31;
  const int l16  = lane & 15;
  const int half = lane >> 4;
  const int blk  = blockIdx.x;                 // 128 blocks
  const int b    = blk >> 4;                   // 16 blocks / batch
  const int q0   = (blk & 15) * 128 + wave * 16;   // first query row (in batch)
  const size_t rowbase = (size_t)b * TSEQ;

  // Q in A-layout registers: lane = M, two K-chunks of 32 (h dim)
  v16bf qa[2];
  {
    const __bf16* qp = Qb + (rowbase + q0 + l16) * HDIM;
    #pragma unroll
    for (int kc = 0; kc < 2; ++kc) {
      v8bf lo = *(const v8bf*)(qp + kc * 32 + half * 16);
      v8bf hi = *(const v8bf*)(qp + kc * 32 + half * 16 + 8);
      #pragma unroll
      for (int e = 0; e < 8; ++e) { qa[kc][e] = lo[e]; qa[kc][8 + e] = hi[e]; }
    }
  }

  float mrow[8], lsum[8];
  v8f o[4] = {};
  #pragma unroll
  for (int r = 0; r < 8; ++r) { mrow[r] = -1e9f; lsum[r] = 0.f; }
  const float qscale = 0.03125f * 1.44269504088896f;  // C^-0.5 * log2(e)

  for (int ks = 0; ks < q0 + 16; ks += 32) {
    // S = Q K^T for a 16(row) x 32(key) tile
    v8f st[2] = {};
    #pragma unroll
    for (int nt = 0; nt < 2; ++nt) {
      const __bf16* kp = Kb + (rowbase + ks + nt * 16 + l16) * HDIM;
      #pragma unroll
      for (int kc = 0; kc < 2; ++kc) {
        v16bf kb = *(const v16bf*)(kp + kc * 32 + half * 16);
        st[nt] = wmma_bf16(qa[kc], kb, st[nt]);
      }
    }
    // scale + causal mask (C layout: col = l16 (+16), row = q0 + r + 8*half)
    #pragma unroll
    for (int nt = 0; nt < 2; ++nt) {
      #pragma unroll
      for (int r = 0; r < 8; ++r) {
        const int key = ks + nt * 16 + l16;
        const int row = q0 + r + 8 * half;
        const float v = st[nt][r] * qscale;
        st[nt][r] = (key <= row) ? v : -1e9f;
      }
    }
    // online softmax per owned row; also stage P (bf16) to LDS
    #pragma unroll
    for (int r = 0; r < 8; ++r) {
      float mx = fmaxf(st[0][r], st[1][r]);
      #pragma unroll
      for (int msk = 8; msk >= 1; msk >>= 1)
        mx = fmaxf(mx, __shfl_xor(mx, msk, 32));
      const float mnew  = fmaxf(mrow[r], mx);
      const float alpha = exp2f(mrow[r] - mnew);
      const float p0 = exp2f(st[0][r] - mnew);
      const float p1 = exp2f(st[1][r] - mnew);
      float ps = p0 + p1;
      #pragma unroll
      for (int msk = 8; msk >= 1; msk >>= 1)
        ps += __shfl_xor(ps, msk, 32);
      lsum[r] = lsum[r] * alpha + ps;
      mrow[r] = mnew;
      #pragma unroll
      for (int hn = 0; hn < 4; ++hn) o[hn][r] *= alpha;
      const int rowIdx = r + 8 * half;
      pbuf[wave][rowIdx * 40 + l16]      = (__bf16)p0;
      pbuf[wave][rowIdx * 40 + 16 + l16] = (__bf16)p1;
    }
    asm volatile("s_wait_dscnt 0" ::: "memory");   // LDS RAW across lanes
    // reload P in A-layout (lane = row, K = key)
    v16bf pa;
    {
      const __bf16* pp = &pbuf[wave][l16 * 40 + half * 16];
      #pragma unroll
      for (int e = 0; e < 8; ++e) { pa[e] = pp[e]; pa[8 + e] = pp[8 + e]; }
    }
    // O += P x V  (B operand from V^T: lane = h col, K = keys, contiguous)
    #pragma unroll
    for (int hn = 0; hn < 4; ++hn) {
      v16bf vb = *(const v16bf*)(Vt + (size_t)b * HDIM * TSEQ +
                                 (size_t)(hn * 16 + l16) * TSEQ + ks + half * 16);
      o[hn] = wmma_bf16(pa, vb, o[hn]);
    }
  }

  // epilogue: normalize and store fp32
  #pragma unroll
  for (int hn = 0; hn < 4; ++hn) {
    #pragma unroll
    for (int r = 0; r < 8; ++r) {
      const size_t orow = rowbase + q0 + r + 8 * half;
      out[orow * HDIM + hn * 16 + l16] = o[hn][r] / lsum[r];
    }
  }
}

// ---------------------------------------------------------------------------
extern "C" void kernel_launch(void* const* d_in, const int* in_sizes, int n_in,
                              void* d_out, int out_size, void* d_ws, size_t ws_size,
                              hipStream_t stream) {
  const float* x  = (const float*)d_in[0];
  const float* Wq = (const float*)d_in[1];
  const float* Wk = (const float*)d_in[2];
  const float* Wv = (const float*)d_in[3];

  char* ws = (char*)d_ws;
  const size_t WT_BYTES = (size_t)3 * HDIM * CDIM * 2;       // 384 KB
  const size_t QK_BYTES = (size_t)NB * TSEQ * HDIM * 2;      // 2 MB each
  __bf16* Wt = (__bf16*)ws;
  __bf16* Qb = (__bf16*)(ws + WT_BYTES);
  __bf16* Kb = (__bf16*)(ws + WT_BYTES + QK_BYTES);
  __bf16* Vt = (__bf16*)(ws + WT_BYTES + 2 * QK_BYTES);
  float* out = (float*)d_out;

  hipLaunchKernelGGL(wconvert, dim3((3 * HDIM * CDIM) / 256), dim3(256), 0, stream,
                     Wq, Wk, Wv, Wt);
  hipLaunchKernelGGL(qkv_proj, dim3((NB * TSEQ) / 16), dim3(128), 0, stream,
                     x, Wt, Qb, Kb, Vt);
  hipLaunchKernelGGL(attn, dim3((NB * TSEQ) / 128), dim3(256), 0, stream,
                     Qb, Kb, Vt, out);
}
